// LSTMStack_83356725280996
// MI455X (gfx1250) — matrix operand order repaired and checked by reading, not compiled, not run on hardware
//
#include <hip/hip_runtime.h>
#include <hip/hip_bf16.h>
#include <cstddef>
#include <cstdint>

// ---------------------------------------------------------------------------
// LSTM stack: B=32, T=512, H=1024, L=2, gate order i,f,g,o
// Single persistent kernel: 64 blocks x 256 threads (512 waves).
// wave -> (kq in {0,1}, mtile in {0,1}, gpair in {0,1}); block -> ntile.
// bf16 WMMA GEMM (K split 2-way), LDS reduction + gate exchange, c state in
// registers, device-wide generation barrier between layer-steps.
// ---------------------------------------------------------------------------

#define LSTM_B 32
#define LSTM_T 512
#define LSTM_H 1024
#define LSTM_L 2
#define LSTM_G (4 * LSTM_H)        // 4096
#define NBLK   64

typedef __attribute__((ext_vector_type(16))) __bf16 v16bf;
typedef __attribute__((ext_vector_type(8)))  __bf16 v8bf;
typedef __attribute__((ext_vector_type(8)))  float  v8f;

// A-fragment (16x32 bf16, MxK): lane L holds row M=L&15; K chunks
// [k0,k0+8) and [k0+16,k0+24), k0=(L>>4)*8.   (ISA 7.12.2)
__device__ __forceinline__ v16bf load_a_frag(const __bf16* __restrict__ rowbase,
                                             int kk, int lk) {
    const __bf16* p = rowbase + kk + (lk << 3);
    v8bf lo = *(const v8bf*)p;
    v8bf hi = *(const v8bf*)(p + 16);
    return __builtin_shufflevector(lo, hi, 0, 1, 2, 3, 4, 5, 6, 7,
                                           8, 9, 10, 11, 12, 13, 14, 15);
}

// B-fragment (32x16 bf16, KxN) from N-major weights: lane L holds col N=L&15;
// K chunks [k0,k0+8) and [k0+16,k0+24), k0=(L>>4)*8 (same as A).
__device__ __forceinline__ v16bf load_b_frag(const __bf16* __restrict__ rowbase,
                                             int kk, int lk) {
    return load_a_frag(rowbase, kk, lk);
}

#define WMMA_BF16(a, b, c) \
    __builtin_amdgcn_wmma_f32_16x16x32_bf16(false, (a), false, (b), (short)0, (c), false, false)

__device__ __forceinline__ float sigmoid_f(float x) {
    return 1.0f / (1.0f + __expf(-x));
}

// ---------------------------------------------------------------------------
// Prep kernels
// ---------------------------------------------------------------------------
__global__ void zero_u32_kernel(uint32_t* __restrict__ p, int n) {
    int i = blockIdx.x * blockDim.x + threadIdx.x;
    if (i < n) p[i] = 0u;
}

__global__ void cvt_x_kernel(const float* __restrict__ x,
                             __bf16* __restrict__ xb, int n) {
    int i = blockIdx.x * blockDim.x + threadIdx.x;
    if (i < n) xb[i] = (__bf16)x[i];
}

// Tiled transpose + f32->bf16: Wt[lm][n][k] = W_m[l][k][n].
__global__ void __launch_bounds__(256)
cvt_w_kernel(const float* __restrict__ Wx, const float* __restrict__ Wh,
             __bf16* __restrict__ Wt) {
    __shared__ float tile[32][33];
    const int lm = blockIdx.z;
    const int l  = lm >> 1, m = lm & 1;
    const float* src = (m ? Wh : Wx) + (size_t)l * LSTM_H * LSTM_G;
    const int n0 = blockIdx.x * 32, k0 = blockIdx.y * 32;
    const int tx = threadIdx.x & 31, ty = threadIdx.x >> 5;   // 32x8
#pragma unroll
    for (int i = 0; i < 4; ++i)
        tile[ty + i * 8][tx] = src[(size_t)(k0 + ty + i * 8) * LSTM_G + n0 + tx];
    __syncthreads();
    __bf16* dst = Wt + (size_t)lm * LSTM_G * LSTM_H;
#pragma unroll
    for (int i = 0; i < 4; ++i)
        dst[(size_t)(n0 + ty + i * 8) * LSTM_H + k0 + tx] = (__bf16)tile[tx][ty + i * 8];
}

// ---------------------------------------------------------------------------
// GEMM: half-K (512) accumulation of two gate tiles, 2-stage pipelined.
// ---------------------------------------------------------------------------
__device__ __forceinline__ void gemm_pair(
    const __bf16* __restrict__ aXbase, const __bf16* __restrict__ aHbase,
    const __bf16* __restrict__ bXA, const __bf16* __restrict__ bXB,
    const __bf16* __restrict__ bHA, const __bf16* __restrict__ bHB,
    int kbase, int lk, v8f& accA, v8f& accB)
{
    v16bf aX0 = load_a_frag(aXbase, kbase, lk);
    v16bf aH0 = load_a_frag(aHbase, kbase, lk);
    v16bf xA0 = load_b_frag(bXA, kbase, lk);
    v16bf xB0 = load_b_frag(bXB, kbase, lk);
    v16bf hA0 = load_b_frag(bHA, kbase, lk);
    v16bf hB0 = load_b_frag(bHB, kbase, lk);

#pragma unroll 2
    for (int kk = kbase; kk < kbase + 512; kk += 64) {
        v16bf aX1 = load_a_frag(aXbase, kk + 32, lk);
        v16bf aH1 = load_a_frag(aHbase, kk + 32, lk);
        v16bf xA1 = load_b_frag(bXA, kk + 32, lk);
        v16bf xB1 = load_b_frag(bXB, kk + 32, lk);
        v16bf hA1 = load_b_frag(bHA, kk + 32, lk);
        v16bf hB1 = load_b_frag(bHB, kk + 32, lk);

        accA = WMMA_BF16(aX0, xA0, accA);
        accA = WMMA_BF16(aH0, hA0, accA);
        accB = WMMA_BF16(aX0, xB0, accB);
        accB = WMMA_BF16(aH0, hB0, accB);

        // final iteration's prefetch is dead but stays inside d_ws
        aX0 = load_a_frag(aXbase, kk + 64, lk);
        aH0 = load_a_frag(aHbase, kk + 64, lk);
        xA0 = load_b_frag(bXA, kk + 64, lk);
        xB0 = load_b_frag(bXB, kk + 64, lk);
        hA0 = load_b_frag(bHA, kk + 64, lk);
        hB0 = load_b_frag(bHB, kk + 64, lk);

        accA = WMMA_BF16(aX1, xA1, accA);
        accA = WMMA_BF16(aH1, hA1, accA);
        accB = WMMA_BF16(aX1, xB1, accB);
        accB = WMMA_BF16(aH1, hB1, accB);
    }
}

// ---------------------------------------------------------------------------
// Fused epilogue: sum K-split partials + gate exchange from LDS, apply
// sigmoid/tanh, update register-resident c, store h (bf16), y / finals (f32).
// ---------------------------------------------------------------------------
__device__ __forceinline__ void lstm_epilogue(
    float (*xg)[2][256], const float* __restrict__ bias,
    int mtile, int m0, int ncol, int lane, int lk,
    float (&cr)[8],
    __bf16* __restrict__ hnext,            // [32][1024] bf16
    float* __restrict__ yrow,              // out + t*H, or nullptr
    float* __restrict__ hf_out,            // out tail (layer block), or nullptr
    float* __restrict__ cf_out)
{
    const int w0 = 2 * mtile;
    v8f zi = *(v8f*)&xg[w0 + 0][0][lane * 8];
    v8f zf = *(v8f*)&xg[w0 + 0][1][lane * 8];
    v8f zg = *(v8f*)&xg[4 + w0 + 0][0][lane * 8];
    v8f zo = *(v8f*)&xg[4 + w0 + 0][1][lane * 8];
    zi += *(v8f*)&xg[w0 + 1][0][lane * 8];
    zf += *(v8f*)&xg[w0 + 1][1][lane * 8];
    zg += *(v8f*)&xg[4 + w0 + 1][0][lane * 8];
    zo += *(v8f*)&xg[4 + w0 + 1][1][lane * 8];

    const float bI = bias[0 * LSTM_H + ncol];
    const float bF = bias[1 * LSTM_H + ncol];
    const float bG = bias[2 * LSTM_H + ncol];
    const float bO = bias[3 * LSTM_H + ncol];

#pragma unroll
    for (int v = 0; v < 8; ++v) {
        const int m = m0 + (lk << 3) + v;          // C/D tile: M = v + 8*(lane>=16)
        const size_t idx = (size_t)m * LSTM_H + ncol;
        const float xi = zi[v] + bI;
        const float xf = zf[v] + bF;
        const float xv = zg[v] + bG;
        const float xo = zo[v] + bO;
        const float cn = sigmoid_f(xf) * cr[v] + sigmoid_f(xi) * tanhf(xv);
        const float hn = sigmoid_f(xo) * tanhf(cn);
        cr[v] = cn;
        hnext[idx] = (__bf16)hn;
        if (yrow)   yrow[(size_t)m * (LSTM_T * LSTM_H) + ncol] = hn;
        if (hf_out) { hf_out[idx] = hn; cf_out[idx] = cn; }
    }
}

// ---------------------------------------------------------------------------
// Persistent LSTM kernel: loops t x layers with a device-wide generation
// barrier (monotonic agent-scope counter, one polling lane per block).
// ---------------------------------------------------------------------------
__global__ void __launch_bounds__(256)
lstm_persistent(const __bf16* __restrict__ xb,      // [32][512][1024] bf16
                __bf16* __restrict__ hbf,           // [2 parity][2 layer][32][1024] bf16
                const __bf16* __restrict__ wt_all,  // [2][2][4096][1024] bf16
                const float* __restrict__ bias_all, // [2][4096]
                float* __restrict__ out,            // y | h_f | c_f
                unsigned* __restrict__ bar)
{
    __shared__ float xg[8][2][256];                 // per-wave partial gate tiles

    const int widx  = threadIdx.x >> 5;             // 0..7
    const int lane  = threadIdx.x & 31;
    const int kq    = widx & 1;
    const int mtile = (widx >> 1) & 1;
    const int gpair = widx >> 2;
    const int ntile = blockIdx.x;                   // 0..63
    const int m0    = mtile << 4;
    const int ncol0 = ntile << 4;
    const int ln    = lane & 15;
    const int lk    = lane >> 4;
    const int g0    = gpair << 1;
    const int kbase = kq << 9;                      // 0 or 512
    const bool epi  = (gpair == 0) && (kq == 0);
    const int HB    = LSTM_B * LSTM_H;              // 32768
    const size_t YOFF = (size_t)LSTM_B * LSTM_T * LSTM_H;

    float c0[8], c1[8];
#pragma unroll
    for (int v = 0; v < 8; ++v) { c0[v] = 0.0f; c1[v] = 0.0f; }

    unsigned barstep = 0;

#pragma unroll 1
    for (int t = 0; t < LSTM_T; ++t) {
        const int pr = t & 1, pw = pr ^ 1;
#pragma unroll 1
        for (int l = 0; l < LSTM_L; ++l) {
            const __bf16* inp_row =
                (l == 0) ? xb + (size_t)(m0 + ln) * (LSTM_T * LSTM_H) + (size_t)t * LSTM_H
                         : hbf + (size_t)(pw * LSTM_L + 0) * HB + (size_t)(m0 + ln) * LSTM_H;
            const __bf16* hprev_row =
                hbf + (size_t)(pr * LSTM_L + l) * HB + (size_t)(m0 + ln) * LSTM_H;
            const __bf16* wt = wt_all + (size_t)l * (2 * LSTM_G * LSTM_H);
            const __bf16* bXA = wt + (size_t)((g0 + 0) * LSTM_H + ncol0 + ln) * LSTM_H;
            const __bf16* bXB = wt + (size_t)((g0 + 1) * LSTM_H + ncol0 + ln) * LSTM_H;
            const __bf16* bHA = wt + (size_t)(LSTM_G + (g0 + 0) * LSTM_H + ncol0 + ln) * LSTM_H;
            const __bf16* bHB = wt + (size_t)(LSTM_G + (g0 + 1) * LSTM_H + ncol0 + ln) * LSTM_H;

            v8f accA = {}, accB = {};
            gemm_pair(inp_row, hprev_row, bXA, bXB, bHA, bHB, kbase, lk, accA, accB);

            *(v8f*)&xg[widx][0][lane * 8] = accA;
            *(v8f*)&xg[widx][1][lane * 8] = accB;
            __syncthreads();

            if (epi) {
                __bf16* hnext = hbf + (size_t)(pw * LSTM_L + l) * HB;
                float* yrow  = (l == 1) ? out + (size_t)t * LSTM_H : (float*)nullptr;
                float* hf_o  = (t == LSTM_T - 1) ? out + YOFF + (size_t)l * HB : (float*)nullptr;
                float* cf_o  = (t == LSTM_T - 1) ? out + YOFF + (size_t)LSTM_L * HB + (size_t)l * HB
                                                 : (float*)nullptr;
                if (l == 0)
                    lstm_epilogue(xg, bias_all, mtile, m0, ncol0 + ln, lane, lk,
                                  c0, hnext, yrow, hf_o, cf_o);
                else
                    lstm_epilogue(xg, bias_all + LSTM_G, mtile, m0, ncol0 + ln, lane, lk,
                                  c1, hnext, yrow, hf_o, cf_o);
            }
            __threadfence();                        // release h stores to agent scope

            // ---- device-wide generation barrier ----
            ++barstep;
            __syncthreads();
            if (threadIdx.x == 0) {
                __hip_atomic_fetch_add(bar, 1u, __ATOMIC_ACQ_REL, __HIP_MEMORY_SCOPE_AGENT);
                const unsigned target = barstep * NBLK;
                while (__hip_atomic_load(bar, __ATOMIC_ACQUIRE, __HIP_MEMORY_SCOPE_AGENT) < target)
                    __builtin_amdgcn_s_sleep(2);
            }
            __syncthreads();
        }
    }
}

// ---------------------------------------------------------------------------
// Host launcher: 4 launches total (zero, cvt_x, cvt_w, persistent LSTM).
// ---------------------------------------------------------------------------
extern "C" void kernel_launch(void* const* d_in, const int* in_sizes, int n_in,
                              void* d_out, int out_size, void* d_ws, size_t ws_size,
                              hipStream_t stream) {
    (void)in_sizes; (void)n_in; (void)out_size; (void)ws_size;

    const float* x  = (const float*)d_in[0];   // [32,512,1024]
    const float* Wx = (const float*)d_in[1];   // [2,1024,4096]
    const float* Wh = (const float*)d_in[2];   // [2,1024,4096]
    const float* bb = (const float*)d_in[3];   // [2,4096]
    float* out = (float*)d_out;                // y | h_f | c_f

    // ---- workspace carve-up (bytes) ----
    char* ws = (char*)d_ws;
    const size_t WT_BYTES = (size_t)LSTM_L * 2 * LSTM_G * LSTM_H * 2; // 32 MB
    const size_t XB_BYTES = (size_t)LSTM_B * LSTM_T * LSTM_H * 2;     // 32 MB
    const size_t HB_BYTES = (size_t)2 * LSTM_L * LSTM_B * LSTM_H * 2; // 256 KB (2 parities)
    const size_t PAD      = 4096;                                     // prefetch overrun pad

    __bf16*   Wt  = (__bf16*)ws;
    __bf16*   xb  = (__bf16*)(ws + WT_BYTES);
    __bf16*   hbf = (__bf16*)(ws + WT_BYTES + XB_BYTES);
    unsigned* bar = (unsigned*)(ws + WT_BYTES + XB_BYTES + HB_BYTES + PAD);

    // ---- zero h state + pad + barrier counter (every call) ----
    {
        int words = (int)((HB_BYTES + PAD + 64) / 4);
        zero_u32_kernel<<<(words + 255) / 256, 256, 0, stream>>>(
            (uint32_t*)(ws + WT_BYTES + XB_BYTES), words);
    }
    // ---- convert x to bf16 (coalesced) ----
    {
        int n = LSTM_B * LSTM_T * LSTM_H;                 // 16777216
        cvt_x_kernel<<<(n + 255) / 256, 256, 0, stream>>>(x, xb, n);
    }
    // ---- tiled transpose + convert weights to N-major bf16 ----
    {
        dim3 g(LSTM_G / 32, LSTM_H / 32, LSTM_L * 2);     // 128 x 32 x 4
        cvt_w_kernel<<<g, 256, 0, stream>>>(Wx, Wh, Wt);
    }
    // ---- persistent recurrence over all 512 timesteps x 2 layers ----
    lstm_persistent<<<NBLK, 256, 0, stream>>>(xb, hbf, Wt, bb, out, bar);
}
